// LSTMCell_48679159332903
// MI455X (gfx1250) — matrix operand order repaired and checked by reading, not compiled, not run on hardware
//
#include <hip/hip_runtime.h>

// CDNA5 (gfx1250) vector types
typedef __attribute__((ext_vector_type(16))) __bf16       v16bf;
typedef __attribute__((ext_vector_type(8)))  float        v8f;
typedef __attribute__((ext_vector_type(4)))  unsigned int v4u;
typedef __attribute__((ext_vector_type(8)))  int          v8i;
typedef __attribute__((ext_vector_type(4)))  int          v4i;

#define DHID   128
#define FOURD  512
#define KTOT   256
// TDM writes 16 bf16 (32B) per row, then pads 16B -> 48B LDS row stride:
// 16B-aligned for ds_load_b128, and stride 12 dwords breaks bank-conflicts.
#define ROW_LDS_B        48
#define GATE_TILE_BYTES  (KTOT * ROW_LDS_B)      // 12288
#define BUF_BYTES        (4 * GATE_TILE_BYTES)   // 49152 per buffer

__device__ __forceinline__ float fast_sigmoid(float x) {
    return 1.0f / (1.0f + __expf(-x));
}
__device__ __forceinline__ float fast_tanh(float x) {
    return 2.0f / (1.0f + __expf(-2.0f * x)) - 1.0f;
}

// ---- TDM: load one 16-col x 256-row bf16 tile of Wc[256][512] into LDS ----
// D# per CDNA5 ISA ch.8: group0 = {flags, lds_addr, global_addr, type},
// group1 = dims/strides/padding. 2D tensor -> groups 2/3 zero (NULL).
__device__ __forceinline__ void tdm_load_tile(unsigned long long gaddr,
                                              unsigned int ldsaddr)
{
    v4u g0;
    g0.x = 1u;                                        // count=1 (valid descriptor)
    g0.y = ldsaddr;                                   // LDS byte address
    g0.z = (unsigned int)gaddr;                       // global_addr[31:0]
    g0.w = (unsigned int)((gaddr >> 32) & 0x01FFFFFFull)
         | (2u << 30);                                // global_addr[56:32] | type=2

    v8i g1;
    g1[0] = (int)((1u << 16)       // data_size = 2 bytes (bf16)
                | (1u << 20)       // pad_enable
                | (2u << 22)       // pad_interval: 8 DWORDs (one 32B row)
                | (3u << 25));     // pad_amount: 4 DWORDs (16B) -> 48B row stride
    g1[1] = (int)(512u << 16);     // tensor_dim0 = 512   (bits 79:48)
    g1[2] = (int)(256u << 16);     // tensor_dim1 = 256   (bits 111:80)
    g1[3] = (int)(16u  << 16);     // tile_dim0   = 16    (bits 127:112)
    g1[4] = 256;                   // tile_dim1   = 256   (bits 143:128)
    g1[5] = 512;                   // tensor_dim0_stride = 512 (bits 207:160)
    g1[6] = 0;
    g1[7] = 2;                     // tensor_dim1_stride = 131072 (bits 255:208)

    v4i z4 = {0, 0, 0, 0};
#if defined(__clang_major__) && (__clang_major__ >= 23)
    v8i z8 = {0, 0, 0, 0, 0, 0, 0, 0};
    __builtin_amdgcn_tensor_load_to_lds(g0, g1, z4, z4, z8, 0);
#else
    __builtin_amdgcn_tensor_load_to_lds(g0, g1, z4, z4, 0);
#endif
}

// ---- One-shot prep: Wc[256][512] = bf16([Uw;Ww]); Bc[512] = Ub+Wb ----------
__global__ __launch_bounds__(256)
void lstm_prep_weights(const float* __restrict__ Uw, const float* __restrict__ Ww,
                       const float* __restrict__ Ub, const float* __restrict__ Wb,
                       __bf16* __restrict__ Wc, float* __restrict__ Bc)
{
    const int idx = blockIdx.x * 256 + threadIdx.x;        // 0..131071
    const float v = (idx < DHID * FOURD) ? Uw[idx] : Ww[idx - DHID * FOURD];
    Wc[idx] = (__bf16)v;
    if (idx < FOURD) Bc[idx] = Ub[idx] + Wb[idx];
}

// ---- Main fused kernel -----------------------------------------------------
// gates = [X|H0] @ Wc + Bc, then LSTM elementwise. 8 waves x 16 rows = 128
// rows/block. TDM double-buffers per-d-group weight tiles into LDS while the
// waves run WMMAs on the previous group.
__global__ __launch_bounds__(256)
void lstm_cell_wmma_tdm(const float* __restrict__ X,
                        const float* __restrict__ C0,
                        const float* __restrict__ H0,
                        const __bf16* __restrict__ Wc,
                        const float* __restrict__ Bc,
                        float* __restrict__ C2,
                        float* __restrict__ H1)
{
    __shared__ __attribute__((aligned(128))) unsigned char wlds[2 * BUF_BYTES];

    const int tid   = threadIdx.x;
    const int lane  = tid & 31;
    const int wave  = tid >> 5;
    const int khalf = lane >> 4;
    const int mrow  = lane & 15;

    const int rowbase = blockIdx.x * 128 + wave * 16;
    const int arow    = rowbase + mrow;

    // A fragments: 16 rows x K=256 bf16, resident in VGPRs (ISA 16-bit A layout).
    v16bf afrag[8];
#pragma unroll
    for (int s = 0; s < 8; ++s) {
#pragma unroll
        for (int run = 0; run < 2; ++run) {
            const int K0 = s * 32 + run * 16 + khalf * 8;
            const float* src = (K0 < DHID) ? (X  + (size_t)arow * DHID + K0)
                                           : (H0 + (size_t)arow * DHID + (K0 - DHID));
            const float4 f0 = *(const float4*)(src);
            const float4 f1 = *(const float4*)(src + 4);
            const int eb = run * 8;
            afrag[s][eb + 0] = (__bf16)f0.x;  afrag[s][eb + 1] = (__bf16)f0.y;
            afrag[s][eb + 2] = (__bf16)f0.z;  afrag[s][eb + 3] = (__bf16)f0.w;
            afrag[s][eb + 4] = (__bf16)f1.x;  afrag[s][eb + 5] = (__bf16)f1.y;
            afrag[s][eb + 6] = (__bf16)f1.z;  afrag[s][eb + 7] = (__bf16)f1.w;
        }
    }

    const unsigned int lds_base =
        (unsigned int)(unsigned long long)(uintptr_t)&wlds[0];
    const unsigned long long wbase = (unsigned long long)(uintptr_t)Wc;

    // Prologue: wave 0 DMAs d-group 0 weight tiles (4 gates) into buffer 0.
    if (wave == 0) {
#pragma unroll
        for (int g = 0; g < 4; ++g)
            tdm_load_tile(wbase + 2ull * (unsigned long long)(g * DHID),
                          lds_base + g * GATE_TILE_BYTES);
    }

    for (int dg = 0; dg < 8; ++dg) {
        const int buf = dg & 1;
        if (wave == 0) {
            if (dg + 1 < 8) {
                const unsigned int nb = (unsigned int)(buf ^ 1) * BUF_BYTES;
#pragma unroll
                for (int g = 0; g < 4; ++g)
                    tdm_load_tile(
                        wbase + 2ull * (unsigned long long)(g * DHID + (dg + 1) * 16),
                        lds_base + nb + g * GATE_TILE_BYTES);
                __builtin_amdgcn_s_wait_tensorcnt(4);   // current group's 4 done
            } else {
                __builtin_amdgcn_s_wait_tensorcnt(0);
            }
        }
        __syncthreads();   // weights for this d-group visible to all waves

        const unsigned char* wb = wlds + buf * BUF_BYTES;

        v8f acc[4];
#pragma unroll
        for (int g = 0; g < 4; ++g)
#pragma unroll
            for (int r = 0; r < 8; ++r) acc[g][r] = 0.0f;

        // Register-double-buffered B fragments: lane L holds column L%16, element e = K-row kbase+16*(e>>3)+8*(L>>4)+(e&7).
        union BU { uint4 u[2]; v16bf v; };
        BU bb[2][4];
#pragma unroll
        for (int g = 0; g < 4; ++g) {
#pragma unroll
            for (int e = 0; e < 16; ++e)
                bb[0][g].v[e] = *(const __bf16*)(wb + g * GATE_TILE_BYTES
                    + ((e >> 3) * 16 + khalf * 8 + (e & 7)) * ROW_LDS_B + mrow * 2);
        }
#pragma unroll
        for (int s = 0; s < 8; ++s) {
            if (s < 7) {
#pragma unroll
                for (int g = 0; g < 4; ++g) {
#pragma unroll
                    for (int e = 0; e < 16; ++e)
                        bb[(s + 1) & 1][g].v[e] = *(const __bf16*)(wb + g * GATE_TILE_BYTES
                            + ((s + 1) * 32 + (e >> 3) * 16 + khalf * 8 + (e & 7)) * ROW_LDS_B + mrow * 2);
                }
            }
#pragma unroll
            for (int g = 0; g < 4; ++g) {
                acc[g] = __builtin_amdgcn_wmma_f32_16x16x32_bf16(
                    false, afrag[s], false, bb[s & 1][g].v,
                    (short)0, acc[g], false, false);
            }
        }

        // LSTM elementwise epilogue; bias folded here (scalar per gate/lane).
        const int colbase = dg * 16 + mrow;
        const float bF = Bc[0 * DHID + colbase];
        const float bC = Bc[1 * DHID + colbase];
        const float bI = Bc[2 * DHID + colbase];
        const float bO = Bc[3 * DHID + colbase];
#pragma unroll
        for (int r = 0; r < 8; ++r) {
            const int row = rowbase + r + 8 * khalf;   // C/D VGPR->M mapping
            const size_t idx = (size_t)row * DHID + colbase;
            const float fv = fast_sigmoid(acc[0][r] + bF);
            const float cv = fast_tanh   (acc[1][r] + bC);
            const float iv = fast_sigmoid(acc[2][r] + bI);
            const float ov = fast_sigmoid(acc[3][r] + bO);
            const float c2 = fv * C0[idx] + iv * cv;
            C2[idx] = c2;
            H1[idx] = ov * fast_tanh(c2);
        }
        __syncthreads();   // all reads done before TDM overwrites this buffer
    }
}

extern "C" void kernel_launch(void* const* d_in, const int* in_sizes, int n_in,
                              void* d_out, int out_size, void* d_ws, size_t ws_size,
                              hipStream_t stream) {
    const float* X  = (const float*)d_in[0];
    const float* C0 = (const float*)d_in[1];
    const float* H0 = (const float*)d_in[2];
    const float* Uw = (const float*)d_in[3];
    const float* Ub = (const float*)d_in[4];
    const float* Ww = (const float*)d_in[5];
    const float* Wb = (const float*)d_in[6];

    // Workspace: bf16 Wc[256][512] (256 KB) then fp32 Bc[512] (2 KB).
    __bf16* Wc = (__bf16*)d_ws;
    float*  Bc = (float*)((char*)d_ws + (size_t)KTOT * FOURD * sizeof(__bf16));

    const int Btot = in_sizes[0] / DHID;          // 131072
    float* C2 = (float*)d_out;                    // outputs: C2 then H1
    float* H1 = C2 + (size_t)Btot * DHID;

    lstm_prep_weights<<<(KTOT * FOURD) / 256, 256, 0, stream>>>(Uw, Ww, Ub, Wb, Wc, Bc);
    lstm_cell_wmma_tdm<<<Btot / 128, 256, 0, stream>>>(X, C0, H0, Wc, Bc, C2, H1);
}
